// SPELModel_66348654788739
// MI455X (gfx1250) — matrix operand order repaired and checked by reading, not compiled, not run on hardware
//
#include <hip/hip_runtime.h>
#include <math.h>

// ---- types matching CDNA5 WMMA builtin signatures ----
typedef __bf16 bf16;
typedef __attribute__((ext_vector_type(16))) __bf16 v16bf;
typedef __attribute__((ext_vector_type(8)))  __bf16 bf16x8;
typedef __attribute__((ext_vector_type(4)))  __bf16 bf16x4;
typedef __attribute__((ext_vector_type(8)))  float  v8f;
typedef __attribute__((ext_vector_type(4)))  float  f32x4;

#define M_TOT 1024      // B*T
#define K_TOT 1024      // H
#define N_TOT 50000     // V
#define BLK_M 128       // 8 waves x 16 rows
#define BLK_N 80        // 5 x 16-col tiles per wave (50000 = 625*80 exactly)
#define TK    32        // K slice per stage (bf16 wmma K)
#define SSTR  40        // LDS row stride in bf16 elems (32 padded -> 40, keeps 16B align)

union FragU { v16bf v; bf16x8 h[2]; };

__device__ __forceinline__ void cvt_store(const f32x4 v, bf16* __restrict__ hp,
                                          bf16* __restrict__ lp) {
    bf16x4 h, l;
    #pragma unroll
    for (int e = 0; e < 4; ++e) {
        float x = v[e];
        bf16 hb = (bf16)x;
        h[e] = hb;
        l[e] = (bf16)(x - (float)hb);
    }
    *(bf16x4*)hp = h;
    *(bf16x4*)lp = l;
}

__global__ __launch_bounds__(256)
void gemm_loss_kernel(const float* __restrict__ hidden,
                      const float* __restrict__ W,
                      const float* __restrict__ amask,
                      float* __restrict__ logits,
                      double* __restrict__ partials)
{
    // Only B goes through LDS (real x8 reuse across waves). A is per-wave private.
    __shared__ bf16 sBhi[BLK_N * SSTR];
    __shared__ bf16 sBlo[BLK_N * SSTR];
    __shared__ float sred[8];

    const int tid  = threadIdx.x;
    const int lane = tid & 31;   // wave32
    const int wave = tid >> 5;   // 0..7
    const int mrow = lane & 15;
    const int hi16 = lane >> 4;  // 0 or 1

    const int mBase = blockIdx.y * BLK_M;
    const int nBase = blockIdx.x * BLK_N;

    // ---- per-lane A pointer: row = own M row; two 8-float k-runs at kb8, kb8+16 ----
    const float* aptr = hidden + (size_t)(mBase + wave * 16 + mrow) * K_TOT + hi16 * 8;

    // ---- per-thread B staging coordinates (fixed across K-steps) ----
    int bRow[3], bC4[3];
    bool bVal[3];
    #pragma unroll
    for (int q = 0; q < 3; ++q) { int li = q * 256 + tid; bRow[q] = li >> 3; bC4[q] = li & 7;
                                  bVal[q] = li < (BLK_N * TK / 4); }

    f32x4 pA[4];   // raw fp32 A fragment (two runs of 8)
    f32x4 pB[3];   // raw fp32 B staging chunk
    auto load_A = [&](int k0) {
        pA[0] = *(const f32x4*)(aptr + k0);
        pA[1] = *(const f32x4*)(aptr + k0 + 4);
        pA[2] = *(const f32x4*)(aptr + k0 + 16);
        pA[3] = *(const f32x4*)(aptr + k0 + 20);
    };
    auto load_B = [&](int k0) {
        #pragma unroll
        for (int q = 0; q < 3; ++q)
            if (bVal[q])
                pB[q] = *(const f32x4*)(W + (size_t)(nBase + bRow[q]) * K_TOT + k0 + bC4[q] * 4);
    };

    v8f acc[5] = {};

    load_A(0);
    load_B(0);
    for (int k0 = 0; k0 < K_TOT; k0 += TK) {
        // ---- stage B slab fp32 -> bf16 hi/lo into LDS ----
        #pragma unroll
        for (int q = 0; q < 3; ++q)
            if (bVal[q])
                cvt_store(pB[q], sBhi + bRow[q] * SSTR + bC4[q] * 4,
                                 sBlo + bRow[q] * SSTR + bC4[q] * 4);

        // ---- build A hi/lo fragments fully in registers (no LDS round-trip) ----
        FragU ahi, alo;
        #pragma unroll
        for (int q = 0; q < 4; ++q) {
            #pragma unroll
            for (int e = 0; e < 4; ++e) {
                float x = pA[q][e];
                bf16 hb = (bf16)x;
                ahi.v[q * 4 + e] = hb;
                alo.v[q * 4 + e] = (bf16)(x - (float)hb);
            }
        }
        __syncthreads();

        // ---- issue next K-step's global loads; waits hide behind wmma below ----
        if (k0 + TK < K_TOT) { load_A(k0 + TK); load_B(k0 + TK); }

        // ---- 5 N-tiles: B fragment (two 8-k runs at 8h, 16+8h, as A) + bf16x3 wmma ----
        #pragma unroll
        for (int t = 0; t < 5; ++t) {
            FragU bhi, blo;
            const int boff = (t * 16 + mrow) * SSTR + hi16 * 8;
            bhi.h[0] = *(const bf16x8*)(sBhi + boff);
            bhi.h[1] = *(const bf16x8*)(sBhi + boff + 16);
            blo.h[0] = *(const bf16x8*)(sBlo + boff);
            blo.h[1] = *(const bf16x8*)(sBlo + boff + 16);

            acc[t] = __builtin_amdgcn_wmma_f32_16x16x32_bf16(
                         false, ahi.v, false, bhi.v, (short)0, acc[t], false, false);
            acc[t] = __builtin_amdgcn_wmma_f32_16x16x32_bf16(
                         false, ahi.v, false, blo.v, (short)0, acc[t], false, false);
            acc[t] = __builtin_amdgcn_wmma_f32_16x16x32_bf16(
                         false, alo.v, false, bhi.v, (short)0, acc[t], false, false);
        }
        __syncthreads();
    }

    // ---- epilogue: store logits (non-temporal) + accumulate base-loss term ----
    float lacc = 0.f;
    const int gRowBase = mBase + wave * 16 + hi16 * 8;
    #pragma unroll
    for (int r = 0; r < 8; ++r) {
        const int gRow = gRowBase + r;
        const float mk = amask[gRow];
        #pragma unroll
        for (int t = 0; t < 5; ++t) {
            const int gCol = nBase + t * 16 + mrow;
            float x = acc[t][r];
            __builtin_nontemporal_store(x, logits + (size_t)gRow * N_TOT + gCol);
            float base = fmaxf(x, 0.f) + log1pf(__expf(-fabsf(x)));
            lacc += base * mk;
        }
    }

    // wave reduce (wave32) then deterministic per-block partial
    #pragma unroll
    for (int off = 16; off > 0; off >>= 1)
        lacc += __shfl_down(lacc, off, 32);
    if (lane == 0) sred[wave] = lacc;
    __syncthreads();
    if (tid == 0) {
        float s = 0.f;
        #pragma unroll
        for (int wv = 0; wv < 8; ++wv) s += sred[wv];
        partials[blockIdx.y * gridDim.x + blockIdx.x] = (double)s;
    }
}

__global__ __launch_bounds__(1024)
void finalize_kernel(const float* __restrict__ logits,
                     const int* __restrict__ labels,
                     const float* __restrict__ amask,
                     const double* __restrict__ partials,
                     int nPartials,
                     float* __restrict__ lossOut)
{
    __shared__ double sb[1024];
    __shared__ double sg[1024];
    __shared__ double sm[1024];
    const int t = threadIdx.x;     // == flattened (b,t) position, B*T == 1024

    double bsum = 0.0;
    for (int i = t; i < nPartials; i += 1024) bsum += partials[i];

    const int   lab = labels[t];
    const float mk  = amask[t];
    const bool  valid = (mk > 0.f) && (lab >= 0);
    const int   cl  = lab < 0 ? 0 : lab;
    const float g   = logits[(size_t)t * N_TOT + cl];

    sb[t] = bsum;
    sg[t] = valid ? (double)g * (double)mk : 0.0;
    sm[t] = (double)mk;
    __syncthreads();
    for (int off = 512; off > 0; off >>= 1) {
        if (t < off) { sb[t] += sb[t + off]; sg[t] += sg[t + off]; sm[t] += sm[t + off]; }
        __syncthreads();
    }
    if (t == 0) lossOut[0] = (float)((sb[0] - sg[0]) / sm[0]);
}

extern "C" void kernel_launch(void* const* d_in, const int* in_sizes, int n_in,
                              void* d_out, int out_size, void* d_ws, size_t ws_size,
                              hipStream_t stream) {
    const float* hidden = (const float*)d_in[0];  // [2,512,1024] fp32
    const float* W      = (const float*)d_in[1];  // [50000,1024] fp32
    const int*   labels = (const int*)d_in[2];    // [2,512] int
    const float* amask  = (const float*)d_in[3];  // [2,512] fp32

    float* logits  = (float*)d_out;                       // [1024, 50000]
    float* lossOut = (float*)d_out + (size_t)M_TOT * N_TOT;
    double* partials = (double*)d_ws;                     // 5000 doubles (40 KB)

    dim3 grid(N_TOT / BLK_N, M_TOT / BLK_M);              // (625, 8)
    gemm_loss_kernel<<<grid, 256, 0, stream>>>(hidden, W, amask, logits, partials);
    finalize_kernel<<<1, 1024, 0, stream>>>(logits, labels, amask, partials,
                                            (N_TOT / BLK_N) * (M_TOT / BLK_M), lossOut);
}
